// MinGRU_11656541241657
// MI455X (gfx1250) — matrix operand order repaired and checked by reading, not compiled, not run on hardware
//
#include <hip/hip_runtime.h>
#include <hip/hip_bf16.h>
#include <math.h>

// Problem constants (from reference): B=8, T=4096, D=1024, H=1024
#define B_  8
#define T_  4096
#define D_  1024
#define H_  1024
#define NC  64          // scan chunks per sequence
#define LC  (T_ / NC)   // 64 steps per chunk

typedef __attribute__((ext_vector_type(16))) __bf16 v16bf;
typedef __attribute__((ext_vector_type(8)))  __bf16 v8bf;
typedef __attribute__((ext_vector_type(4)))  __bf16 v4bf;
typedef __attribute__((ext_vector_type(8)))  float  v8f;

#define WMMA_BF16(A, Bm, C) \
    __builtin_amdgcn_wmma_f32_16x16x32_bf16(false, (A), false, (Bm), (short)0, (C), false, false)

// ---------------- math helpers ----------------
__device__ __forceinline__ float softplus_f(float x) {
    return fmaxf(x, 0.0f) + log1pf(__expf(-fabsf(x)));   // overflow-safe log(1+e^x)
}
__device__ __forceinline__ float log_g_f(float x) {
    return (x >= 0.0f) ? __logf(x + 0.5f) : -softplus_f(-x);
}
__device__ __forceinline__ float logaddexp_f(float a, float b) {
    float m = fmaxf(a, b);
    float r = m + log1pf(__expf(fminf(a, b) - m));
    return (m == -INFINITY) ? -INFINITY : r;   // guards NaN from (-inf)-(-inf)
}

// ---------------------------------------------------------------------------
// One-shot fp32 -> bf16 hi/lo split (a ~= hi + lo). Runs once per array so
// the GEMM inner loop carries zero conversion VALU.
// ---------------------------------------------------------------------------
__global__ void __launch_bounds__(256)
convert_hilo(const float* __restrict__ src, __bf16* __restrict__ hi,
             __bf16* __restrict__ lo, int n4)
{
    const int i = blockIdx.x * 256 + threadIdx.x;
    if (i >= n4) return;
    const float4 f = ((const float4*)src)[i];
    float v[4] = {f.x, f.y, f.z, f.w};
    v4bf h, l;
#pragma unroll
    for (int j = 0; j < 4; ++j) {
        __bf16 hb = (__bf16)v[j];
        h[j] = hb;
        l[j] = (__bf16)(v[j] - (float)hb);
    }
    ((v4bf*)hi)[i] = h;
    ((v4bf*)lo)[i] = l;
}

// assemble a v16bf WMMA fragment from two aligned 16B chunks
__device__ __forceinline__ v16bf ld_frag(const __bf16* p0, const __bf16* p1) {
    v8bf a = *(const v8bf*)p0;
    v8bf b = *(const v8bf*)p1;
    return __builtin_shufflevector(a, b, 0,1,2,3,4,5,6,7,8,9,10,11,12,13,14,15);
}

// ---------------------------------------------------------------------------
// Dual GEMM (k = x*Wz^T, ph = x*Wh^T) from pre-split bf16, fused log epilogue.
// One wave computes a 32x16 strip of BOTH outputs (2 M-tiles x 1 N-tile):
// B fragments are reused across both M-tiles; x_hi/x_lo (134 MB) stay L2-
// resident so tile re-reads never hit HBM. 12 WMMAs round-robin over 4
// independent accumulators to avoid WMMA->WMMA hazard stalls.
// ---------------------------------------------------------------------------
__global__ void __launch_bounds__(256)
mingru_gemm_act(const __bf16* __restrict__ x_hi, const __bf16* __restrict__ x_lo,
                const __bf16* __restrict__ wz_hi, const __bf16* __restrict__ wz_lo,
                const __bf16* __restrict__ wh_hi, const __bf16* __restrict__ wh_lo,
                const float* __restrict__ bz, const float* __restrict__ bh,
                float* __restrict__ lc_ws, float* __restrict__ lv_ws)
{
    const int lane   = threadIdx.x & 31;
    const int warp   = threadIdx.x >> 5;
    const int wave   = blockIdx.x * 8 + warp;        // 65536 waves total
    const int tile_n = wave & (H_ / 16 - 1);         // 64 N-tiles
    const int mp     = wave >> 6;                    // 1024 M-tile pairs

    const int mrow0  = (mp * 32) + (lane & 15);      // A tile 0: lane -> M row
    const int ncol   = tile_n * 16 + (lane & 15);    // B: lane -> N col
    const int a_koff = (lane >> 4) << 3;             // A lane-half K offset (0 / 8)
    const int b_koff = (lane >> 4) << 3;             // B lane-half K offset (0 / 8)

    const __bf16* xh0 = x_hi + (size_t)mrow0 * D_ + a_koff;
    const __bf16* xl0 = x_lo + (size_t)mrow0 * D_ + a_koff;
    const __bf16* xh1 = xh0 + (size_t)16 * D_;
    const __bf16* xl1 = xl0 + (size_t)16 * D_;
    const __bf16* wzh = wz_hi + (size_t)ncol * D_ + b_koff;  // W row-major == B^T
    const __bf16* wzl = wz_lo + (size_t)ncol * D_ + b_koff;
    const __bf16* whh = wh_hi + (size_t)ncol * D_ + b_koff;
    const __bf16* whl = wh_lo + (size_t)ncol * D_ + b_koff;

    v8f acc_z0 = {}, acc_h0 = {}, acc_z1 = {}, acc_h1 = {};

    for (int k0 = 0; k0 < D_; k0 += 32) {
        // A fragments (16-bit A 16x32 layout: K = base..base+7, base+16..base+23)
        const v16bf a0h = ld_frag(xh0 + k0, xh0 + k0 + 16);
        const v16bf a0l = ld_frag(xl0 + k0, xl0 + k0 + 16);
        const v16bf a1h = ld_frag(xh1 + k0, xh1 + k0 + 16);
        const v16bf a1l = ld_frag(xl1 + k0, xl1 + k0 + 16);
        // B fragments (16-bit B 32x16 layout: K = base..base+7, base+16..base+23)
        const v16bf bzh = ld_frag(wzh + k0, wzh + k0 + 16);
        const v16bf bzl = ld_frag(wzl + k0, wzl + k0 + 16);
        const v16bf bhh = ld_frag(whh + k0, whh + k0 + 16);
        const v16bf bhl = ld_frag(whl + k0, whl + k0 + 16);

        // 3-term bf16 emulation of the fp32 product (hi*hi + lo*hi + hi*lo),
        // interleaved across 4 independent accumulators
        acc_z0 = WMMA_BF16(a0h, bzh, acc_z0);
        acc_h0 = WMMA_BF16(a0h, bhh, acc_h0);
        acc_z1 = WMMA_BF16(a1h, bzh, acc_z1);
        acc_h1 = WMMA_BF16(a1h, bhh, acc_h1);
        acc_z0 = WMMA_BF16(a0l, bzh, acc_z0);
        acc_h0 = WMMA_BF16(a0l, bhh, acc_h0);
        acc_z1 = WMMA_BF16(a1l, bzh, acc_z1);
        acc_h1 = WMMA_BF16(a1l, bhh, acc_h1);
        acc_z0 = WMMA_BF16(a0h, bzl, acc_z0);
        acc_h0 = WMMA_BF16(a0h, bhl, acc_h0);
        acc_z1 = WMMA_BF16(a1h, bzl, acc_z1);
        acc_h1 = WMMA_BF16(a1h, bhl, acc_h1);
    }

    // fused epilogue: bias + log-space activations
    // C/D layout: VGPR r -> M = base + r + 8*(lane>=16), N = ncol
    const float bzn = bz[ncol];
    const float bhn = bh[ncol];
    const int   mhi = (lane >> 4) << 3;
#pragma unroll
    for (int r = 0; r < 8; ++r) {
#pragma unroll
        for (int tile = 0; tile < 2; ++tile) {
            const int   M  = mp * 32 + tile * 16 + r + mhi;
            const float k  = (tile ? acc_z1[r] : acc_z0[r]) + bzn;
            const float ph = (tile ? acc_h1[r] : acc_h0[r]) + bhn;
            const float lc = -softplus_f(k);                  // log(1 - sigmoid(k))
            const float lv = -softplus_f(-k) + log_g_f(ph);   // log sig(k) + log g(.)
            const size_t idx = (size_t)M * H_ + ncol;
            lc_ws[idx] = lc;
            lv_ws[idx] = lv;
        }
    }
}

// ---------------------------------------------------------------------------
// Chunked associative scan of  s <- logaddexp(lc + s, lv)
// Composition: (A1,V1) then (A2,V2) = (A1+A2, lae(A2+V1, V2)).
// ---------------------------------------------------------------------------
__global__ void __launch_bounds__(256)
mingru_scan_phase1(const float* __restrict__ lc_ws, const float* __restrict__ lv_ws,
                   float* __restrict__ cA, float* __restrict__ cV)
{
    const int g = blockIdx.x * 256 + threadIdx.x;   // B*NC*H threads
    const int h = g & (H_ - 1);
    const int c = (g >> 10) & (NC - 1);
    const int b = g >> 16;
    const size_t base = ((size_t)(b * T_ + c * LC)) * H_ + h;

    float A = 0.0f, V = -INFINITY;
#pragma unroll 8
    for (int t = 0; t < LC; ++t) {
        const float lc = lc_ws[base + (size_t)t * H_];
        const float lv = lv_ws[base + (size_t)t * H_];
        V = logaddexp_f(lc + V, lv);
        A += lc;
    }
    const size_t ci = (size_t)(b * NC + c) * H_ + h;
    cA[ci] = A;
    cV[ci] = V;
}

__global__ void __launch_bounds__(256)
mingru_scan_phase2(const float* __restrict__ h0,
                   const float* __restrict__ cA, const float* __restrict__ cV,
                   float* __restrict__ cS)
{
    const int g = blockIdx.x * 256 + threadIdx.x;   // B*H threads
    const int h = g & (H_ - 1);
    const int b = g >> 10;

    float s = log_g_f(h0[(size_t)b * H_ + h]);      // log_h before t=0
    for (int c = 0; c < NC; ++c) {
        const size_t ci = (size_t)(b * NC + c) * H_ + h;
        cS[ci] = s;                                 // state entering chunk c
        s = logaddexp_f(cA[ci] + s, cV[ci]);
    }
}

__global__ void __launch_bounds__(256)
mingru_scan_phase3(const float* __restrict__ lc_ws, const float* __restrict__ lv_ws,
                   const float* __restrict__ cS, float* __restrict__ out)
{
    const int g = blockIdx.x * 256 + threadIdx.x;   // B*NC*H threads
    const int h = g & (H_ - 1);
    const int c = (g >> 10) & (NC - 1);
    const int b = g >> 16;
    const size_t base = ((size_t)(b * T_ + c * LC)) * H_ + h;

    float s = cS[(size_t)(b * NC + c) * H_ + h];
#pragma unroll 8
    for (int t = 0; t < LC; ++t) {
        const float lc = lc_ws[base + (size_t)t * H_];
        const float lv = lv_ws[base + (size_t)t * H_];
        s = logaddexp_f(lc + s, lv);
        out[base + (size_t)t * H_] = __expf(s);
    }
}

// ---------------------------------------------------------------------------
extern "C" void kernel_launch(void* const* d_in, const int* in_sizes, int n_in,
                              void* d_out, int out_size, void* d_ws, size_t ws_size,
                              hipStream_t stream)
{
    const float* x  = (const float*)d_in[0];
    const float* h0 = (const float*)d_in[1];
    const float* Wz = (const float*)d_in[2];
    const float* bz = (const float*)d_in[3];
    const float* Wh = (const float*)d_in[4];
    const float* bh = (const float*)d_in[5];
    float* out = (float*)d_out;

    // workspace layout (needs ~420 MB):
    const size_t nBTH = (size_t)B_ * T_ * H_;       // 33.5M (also B*T*D)
    const size_t nHD  = (size_t)H_ * D_;            // 1M
    const size_t nBCH = (size_t)B_ * NC * H_;       // 524288
    float*  lc    = (float*)d_ws;
    float*  lv    = lc + nBTH;
    float*  cA    = lv + nBTH;
    float*  cV    = cA + nBCH;
    float*  cS    = cV + nBCH;
    __bf16* x_hi  = (__bf16*)(cS + nBCH);
    __bf16* x_lo  = x_hi + nBTH;
    __bf16* wz_hi = x_lo + nBTH;
    __bf16* wz_lo = wz_hi + nHD;
    __bf16* wh_hi = wz_lo + nHD;
    __bf16* wh_lo = wh_hi + nHD;

    // one-shot bf16 hi/lo splits
    const int n4x = (int)(nBTH / 4);                // 8388608
    const int n4w = (int)(nHD / 4);                 // 262144
    convert_hilo<<<(n4x + 255) / 256, 256, 0, stream>>>(x,  x_hi,  x_lo,  n4x);
    convert_hilo<<<(n4w + 255) / 256, 256, 0, stream>>>(Wz, wz_hi, wz_lo, n4w);
    convert_hilo<<<(n4w + 255) / 256, 256, 0, stream>>>(Wh, wh_hi, wh_lo, n4w);

    // dual GEMM + fused activations (32x16 strip per wave, 8 waves/block)
    const int gemm_blocks = ((B_ * T_ / 32) * (H_ / 16)) / 8;   // 8192
    mingru_gemm_act<<<gemm_blocks, 256, 0, stream>>>(
        x_hi, x_lo, wz_hi, wz_lo, wh_hi, wh_lo, bz, bh, lc, lv);

    // chunked log-space scan
    const int p13_blocks = (B_ * NC * H_) / 256;                // 2048
    mingru_scan_phase1<<<p13_blocks, 256, 0, stream>>>(lc, lv, cA, cV);
    mingru_scan_phase2<<<(B_ * H_) / 256, 256, 0, stream>>>(h0, cA, cV, cS);
    mingru_scan_phase3<<<p13_blocks, 256, 0, stream>>>(lc, lv, cS, out);
}
